// VQ_25357486916144
// MI455X (gfx1250) — matrix operand order repaired and checked by reading, not compiled, not run on hardware
//
#include <hip/hip_runtime.h>
#include <hip/hip_bf16.h>

typedef __attribute__((ext_vector_type(16))) _Float16 v16h;
typedef __attribute__((ext_vector_type(8)))  _Float16 v8h;
typedef __attribute__((ext_vector_type(8)))  float    v8f;

#define VQ_B 16
#define VQ_D 128
#define VQ_N 4096
#define VQ_K 1024

// Row stride (in halves) for staged codebook chunk in LDS: 128 data + 8 pad
// -> 272B row stride shifts banks by 4 per row; 16 rows cover all 64 banks.
#define AROW 136

// ---------------------------------------------------------------------------
// Kernel 1: precompute f16 codebook copy + (-0.5 * row squared-norm) into ws.
// grid = K blocks, 128 threads (one block per codebook row).
// ---------------------------------------------------------------------------
__global__ void vq_prep_kernel(const float* __restrict__ emb,
                               float* __restrict__ emb_nh,
                               _Float16* __restrict__ embh) {
    const int k = blockIdx.x;
    const int d = threadIdx.x;
    const float v = emb[k * VQ_D + d];
    embh[k * VQ_D + d] = (_Float16)v;

    float s = v * v;
    #pragma unroll
    for (int off = 16; off > 0; off >>= 1)
        s += __shfl_xor(s, off, 32);

    __shared__ float ps[4];
    if ((threadIdx.x & 31) == 0) ps[threadIdx.x >> 5] = s;
    __syncthreads();
    if (threadIdx.x == 0)
        emb_nh[k] = -0.5f * (ps[0] + ps[1] + ps[2] + ps[3]);
}

// ---------------------------------------------------------------------------
// Kernel 2: fused GEMM (WMMA f16 -> f32) + argmin + codebook gather.
// grid = (N/256, B), block = 256 threads (8 waves, wave32).
// score[k,n] = dot(ze_n, emb_k) - 0.5*||emb_k||^2 ; argmax_k == argmin d2.
// Each half-lane keeps its own running (max,idx); halves merge once at end.
// ---------------------------------------------------------------------------
__global__ void __launch_bounds__(256)
vq_main_kernel(const float* __restrict__ ze,
               const float* __restrict__ emb,
               const float* __restrict__ emb_nh,
               const _Float16* __restrict__ embh,
               float* __restrict__ out) {
    const int n0   = blockIdx.x * 256;
    const int b    = blockIdx.y;
    const int lane = threadIdx.x & 31;
    const int w    = threadIdx.x >> 5;
    const int half = lane >> 4;     // 0: lanes 0-15, 1: lanes 16-31
    const int col  = lane & 15;

    __shared__ _Float16 sA[2][16 * AROW];
    __shared__ int      sIdx[256];

    // ---- load ze B-fragments (32x16 f16 per d-chunk per tile), keep in regs.
    // B layout (32x16, 16-bit): lanes 0-15 hold K rows {0..7,16..23} (col = lane),
    // lanes 16-31 hold K rows {8..15,24..31}.
    v16h bfrag[4][2];
    #pragma unroll
    for (int tt = 0; tt < 2; ++tt) {
        const int n = n0 + w * 32 + tt * 16 + col;
        const float* zp = ze + (size_t)b * (VQ_D * VQ_N) + n;
        #pragma unroll
        for (int dc = 0; dc < 4; ++dc) {
            v16h f;
            #pragma unroll
            for (int h = 0; h < 16; ++h) {
                const int d = dc * 32 + (h < 8 ? half * 8 + h : 16 + half * 8 + (h - 8));
                f[h] = (_Float16)zp[(size_t)d * VQ_N];
            }
            bfrag[dc][tt] = f;
        }
    }

    // ---- stage codebook chunk 0 into LDS (16 rows x 128 halves, padded).
    {
        const int row = threadIdx.x >> 4;
        const int c8  = threadIdx.x & 15;
        v8h t = *(const v8h*)(embh + row * VQ_D + c8 * 8);
        *(v8h*)(&sA[0][row * AROW + c8 * 8]) = t;
    }
    __syncthreads();

    // per-half-lane running best (rows kc + half*8 + 0..7 across all chunks)
    float runv0 = -3.0e38f, runv1 = -3.0e38f;
    int   runi0 = 0,        runi1 = 0;

    for (int c = 0; c < VQ_K / 16; ++c) {
        const int kc  = c * 16;
        const int row = threadIdx.x >> 4;
        const int c8  = threadIdx.x & 15;

        // prefetch next codebook chunk to registers (overlap with compute)
        v8h nxt;
        if (c + 1 < VQ_K / 16)
            nxt = *(const v8h*)(embh + (size_t)(kc + 16) * VQ_D + row * VQ_D + c8 * 8);

        // C init = -0.5*||emb_k||^2 (pre-scaled in ws): raw b128 pair.
        const float4 q0 = *(const float4*)(emb_nh + kc + half * 8);
        const float4 q1 = *(const float4*)(emb_nh + kc + half * 8 + 4);
        v8f acc0;
        acc0[0] = q0.x; acc0[1] = q0.y; acc0[2] = q0.z; acc0[3] = q0.w;
        acc0[4] = q1.x; acc0[5] = q1.y; acc0[6] = q1.z; acc0[7] = q1.w;
        v8f acc1 = acc0;

        // A layout (16x32, 16-bit): lane m=col; lanes 0-15 hold K {0..7,16..23},
        // lanes 16-31 hold K {8..15,24..31} -> two 16B LDS reads, 32B apart.
        const _Float16* ab = &sA[c & 1][0];
        #pragma unroll
        for (int dc = 0; dc < 4; ++dc) {
            const _Float16* p = ab + col * AROW + dc * 32 + half * 8;
            v8h alo = *(const v8h*)p;
            v8h ahi = *(const v8h*)(p + 16);
            v16h A = __builtin_shufflevector(alo, ahi,
                        0, 1, 2, 3, 4, 5, 6, 7, 8, 9, 10, 11, 12, 13, 14, 15);
            acc0 = __builtin_amdgcn_wmma_f32_16x16x32_f16(
                       false, A, false, bfrag[dc][0], (short)0, acc0, false, false);
            acc1 = __builtin_amdgcn_wmma_f32_16x16x32_f16(
                       false, A, false, bfrag[dc][1], (short)0, acc1, false, false);
        }

        // ---- branchless per-chunk update, half-lanes independent.
        // Ascending row order + strict '>' keeps the lowest index on ties.
        const int base = kc + half * 8;
        {
            float bv = acc0[0]; int bi = base;
            #pragma unroll
            for (int r = 1; r < 8; ++r) {
                const bool g = acc0[r] > bv;
                bv = g ? acc0[r] : bv;
                bi = g ? base + r : bi;
            }
            const bool g2 = bv > runv0;
            runv0 = g2 ? bv : runv0;
            runi0 = g2 ? bi : runi0;
        }
        {
            float bv = acc1[0]; int bi = base;
            #pragma unroll
            for (int r = 1; r < 8; ++r) {
                const bool g = acc1[r] > bv;
                bv = g ? acc1[r] : bv;
                bi = g ? base + r : bi;
            }
            const bool g2 = bv > runv1;
            runv1 = g2 ? bv : runv1;
            runi1 = g2 ? bi : runi1;
        }

        // commit prefetched chunk into the other LDS buffer
        if (c + 1 < VQ_K / 16)
            *(v8h*)(&sA[(c + 1) & 1][row * AROW + c8 * 8]) = nxt;
        __syncthreads();
    }

    // ---- single cross-half combine (tie -> lower global index)
    {
        const float ov0 = __shfl_xor(runv0, 16, 32);
        const int   oi0 = __shfl_xor(runi0, 16, 32);
        const bool  t0  = (ov0 > runv0) || (ov0 == runv0 && oi0 < runi0);
        runv0 = t0 ? ov0 : runv0;
        runi0 = t0 ? oi0 : runi0;

        const float ov1 = __shfl_xor(runv1, 16, 32);
        const int   oi1 = __shfl_xor(runi1, 16, 32);
        const bool  t1  = (ov1 > runv1) || (ov1 == runv1 && oi1 < runi1);
        runv1 = t1 ? ov1 : runv1;
        runi1 = t1 ? oi1 : runi1;
    }

    // ---- publish winning indices
    if (lane < 16) {
        sIdx[w * 32 + col]      = runi0;
        sIdx[w * 32 + 16 + col] = runi1;
    }
    __syncthreads();

    // ---- gather fp32 codebook rows, coalesced along n (contiguous dim)
    {
        const int j   = threadIdx.x;
        const int idx = sIdx[j];
        const float* er = emb + (size_t)idx * VQ_D;
        float* op = out + (size_t)b * (VQ_D * VQ_N) + n0 + j;
        #pragma unroll 4
        for (int d = 0; d < VQ_D; ++d)
            op[(size_t)d * VQ_N] = er[d];
    }
}

// ---------------------------------------------------------------------------
extern "C" void kernel_launch(void* const* d_in, const int* in_sizes, int n_in,
                              void* d_out, int out_size, void* d_ws, size_t ws_size,
                              hipStream_t stream) {
    const float* ze  = (const float*)d_in[0];   // (B, D, N) fp32
    const float* emb = (const float*)d_in[1];   // (K, D)    fp32
    float* out = (float*)d_out;                 // (B, D, N) fp32

    // workspace layout: [0, 4KB) emb_nh (K floats); [4KB, 4KB+256KB) emb f16
    float*    emb_nh = (float*)d_ws;
    _Float16* embh   = (_Float16*)((char*)d_ws + 4096);

    vq_prep_kernel<<<VQ_K, VQ_D, 0, stream>>>(emb, emb_nh, embh);

    dim3 grid(VQ_N / 256, VQ_B);
    vq_main_kernel<<<grid, 256, 0, stream>>>(ze, emb, emb_nh, embh, out);
}
